// EPRInformedGAT_80212809220543
// MI455X (gfx1250) — compile-verified, matrix operand order checked by reading, not run on hardware
//
#include <hip/hip_runtime.h>

// ---------------------------------------------------------------------------
// EPR-informed GATv2 on gfx1250 (MI455X).
// Dominant FLOPs: edge-feature GEMMs [170000x388]x[388x{128,512}] -> WMMA bf16.
// GEMM staging is fully b128-vectorized: A uses a row-guarded uint4 load
// (K-overread is safe: B tile zero-fills k>=K, killing those contributions,
// and the A buffers are padded/inside a large workspace), B uses k-guarded
// float4 loads with on-the-fly bf16 conversion.
// ---------------------------------------------------------------------------

typedef __attribute__((ext_vector_type(16))) __bf16 v16bf;
typedef __attribute__((ext_vector_type(8)))  float  v8f;

union FragBF { v16bf v; uint4 q[2]; };

__device__ __forceinline__ unsigned short f2bf(float f) {
  unsigned int u = __float_as_uint(f);
  unsigned int r = ((u >> 16) & 1u) + 0x7FFFu;   // round-to-nearest-even
  return (unsigned short)((u + r) >> 16);
}

// -------------------------- WMMA GEMM (bf16 -> f32) -------------------------
// C[M,N] = A_bf16[M,K] * B_f32[K,N] (+ bias[N]).
// Block: 256 thr = 8 waves, tile 64x64. Waves form a 4x2 grid; each wave owns
// a 16x32 strip -> 2 WMMA per K-chunk with the A fragment reused.
// Caller guarantees: A base and lda*2 are 16B-aligned (lda in {128,384,392}),
// and rows of A may be safely over-read to the next KC boundary.
#define BM 64
#define BN 64
#define KC 32
#define LDS_STRIDE 40   // halfwords; rows stay 16B aligned (40*2=80 bytes)

__global__ __launch_bounds__(256)
void gemm_bf16_wmma(const unsigned short* __restrict__ A, int lda,
                    const float* __restrict__ B, int ldb,
                    const float* __restrict__ bias,
                    float* __restrict__ C, int ldc,
                    int M, int N, int K) {
  __shared__ unsigned short As[BM * LDS_STRIDE];
  __shared__ unsigned short Bs[BN * LDS_STRIDE];   // stored transposed: [n][k]

  const int tid  = threadIdx.x;
  const int wave = tid >> 5;
  const int lane = tid & 31;
  const int wm = wave & 3;                  // 0..3  (row strip)
  const int wn = wave >> 2;                 // 0..1  (col strip of 32)
  const int r  = lane & 15;
  const int khalf = (lane >> 4) << 3;       // 0 or 8 (ISA 16-bit A layout)

  const int tileM = blockIdx.y * BM;
  const int tileN = blockIdx.x * BN;
  const int colg0 = tileN + wn * 32 + r;
  const int colg1 = colg0 + 16;

  v8f acc0, acc1;
  float bv0 = (bias != nullptr && colg0 < N) ? bias[colg0] : 0.0f;
  float bv1 = (bias != nullptr && colg1 < N) ? bias[colg1] : 0.0f;
#pragma unroll
  for (int i = 0; i < 8; ++i) { acc0[i] = bv0; acc1[i] = bv1; }

  // per-thread staging coordinates
  const int ar = tid >> 2;                  // 0..63 (A row)
  const int ak = (tid & 3) << 3;            // 0,8,16,24 (A k-group)
  const bool arow_ok = (tileM + ar) < M;
  const unsigned short* Arow = A + (size_t)(tileM + ar) * lda + ak;

  for (int k0 = 0; k0 < K; k0 += KC) {
    // ---- stage A tile (BM x KC halfwords): one b128 load+store per thread.
    // No k-guard needed: B zero-fills k>=K, so over-read values are inert.
    uint4 va = {0u, 0u, 0u, 0u};
    if (arow_ok) va = *reinterpret_cast<const uint4*>(Arow + k0);
    *reinterpret_cast<uint4*>(&As[ar * LDS_STRIDE + ak]) = va;

    // ---- stage B tile transposed (KC x BN -> Bs[n][k]), float4 + cvt ----
#pragma unroll
    for (int i = 0; i < 2; ++i) {
      int idx = tid + i * 256;              // 0..511 float4 groups
      int bk = idx >> 4;                    // 0..31
      int bn = (idx & 15) << 2;             // 0..60
      float4 f = {0.0f, 0.0f, 0.0f, 0.0f};
      if (k0 + bk < K)
        f = *reinterpret_cast<const float4*>(&B[(size_t)(k0 + bk) * ldb + tileN + bn]);
      Bs[(bn + 0) * LDS_STRIDE + bk] = f2bf(f.x);
      Bs[(bn + 1) * LDS_STRIDE + bk] = f2bf(f.y);
      Bs[(bn + 2) * LDS_STRIDE + bk] = f2bf(f.z);
      Bs[(bn + 3) * LDS_STRIDE + bk] = f2bf(f.w);
    }

    if (k0 + KC < K) {
      // hint next A tile into cache (global_prefetch_b8)
      __builtin_prefetch(Arow + k0 + KC, 0, 0);
    }
    __syncthreads();

    FragBF fa, fb0, fb1;
    const unsigned short* arow  = &As[(wm * 16 + r) * LDS_STRIDE];
    const unsigned short* brow0 = &Bs[(wn * 32 + r) * LDS_STRIDE];
    const unsigned short* brow1 = brow0 + 16 * LDS_STRIDE;
    // elements 0..7 -> K = khalf..khalf+7 ; elements 8..15 -> K = khalf+16..+23
    fa.q[0]  = *reinterpret_cast<const uint4*>(arow  + khalf);
    fa.q[1]  = *reinterpret_cast<const uint4*>(arow  + khalf + 16);
    fb0.q[0] = *reinterpret_cast<const uint4*>(brow0 + khalf);
    fb0.q[1] = *reinterpret_cast<const uint4*>(brow0 + khalf + 16);
    fb1.q[0] = *reinterpret_cast<const uint4*>(brow1 + khalf);
    fb1.q[1] = *reinterpret_cast<const uint4*>(brow1 + khalf + 16);

    acc0 = __builtin_amdgcn_wmma_f32_16x16x32_bf16(
        false, fa.v, false, fb0.v, (short)0, acc0, false, false);
    acc1 = __builtin_amdgcn_wmma_f32_16x16x32_bf16(
        false, fa.v, false, fb1.v, (short)0, acc1, false, false);
    __syncthreads();
  }

#pragma unroll
  for (int i = 0; i < 8; ++i) {
    int rowg = tileM + wm * 16 + ((lane >> 4) << 3) + i;
    if (rowg < M) {
      if (colg0 < N) C[(size_t)rowg * ldc + colg0] = acc0[i];
      if (colg1 < N) C[(size_t)rowg * ldc + colg1] = acc1[i];
    }
  }
}

// ------------------------------ small kernels -------------------------------

__global__ void zero_f32(float* __restrict__ p, long long n) {
  long long i = (long long)blockIdx.x * blockDim.x + threadIdx.x;
  if (i < n) p[i] = 0.0f;
}

__global__ void cvt_bf16(const float* __restrict__ in, unsigned short* __restrict__ out,
                         long long n) {
  long long i = (long long)blockIdx.x * blockDim.x + threadIdx.x;
  if (i < n) out[i] = f2bf(in[i]);
}

__global__ void epr_mlp(const float* __restrict__ s, const float* __restrict__ W1,
                        const float* __restrict__ b1, const float* __restrict__ W2,
                        const float* __restrict__ b2, float* __restrict__ out, int E) {
  int e = blockIdx.x * blockDim.x + threadIdx.x;
  if (e >= E) return;
  float x = s[e];
  float h[32];
#pragma unroll
  for (int i = 0; i < 32; ++i) {
    float v = x * W1[i] + b1[i];
    h[i] = v > 0.0f ? v : 0.0f;
  }
#pragma unroll
  for (int j = 0; j < 4; ++j) {
    float a = b2[j];
#pragma unroll
    for (int i = 0; i < 32; ++i) a += h[i] * W2[i * 4 + j];
    out[(size_t)e * 4 + j] = a;
  }
}

__global__ void count_dst(const int* __restrict__ dst, float* __restrict__ cnt, int E) {
  int e = blockIdx.x * blockDim.x + threadIdx.x;
  if (e < E) atomicAdd(&cnt[dst[e]], 1.0f);
}

// eattr (bf16, padded stride) + segment-sum for self-loop mean rows
// D = padded stride (392); valid feature dims: [0,384) edge_attr, [384,388) epr
__global__ void build_eattr(const float* __restrict__ edge_attr,
                            const float* __restrict__ epr_feat,
                            const int* __restrict__ dst,
                            unsigned short* __restrict__ eattr_bf,
                            float* __restrict__ selfsum, int E, int D) {
  long long i = (long long)blockIdx.x * blockDim.x + threadIdx.x;
  if (i >= (long long)E * D) return;
  int e = (int)(i / D), d = (int)(i % D);
  float v = 0.0f;
  if (d < 384)      v = edge_attr[(size_t)e * 384 + d];
  else if (d < 388) v = epr_feat[(size_t)e * 4 + (d - 384)];
  eattr_bf[i] = f2bf(v);
  if (d < 388) atomicAdd(&selfsum[(size_t)dst[e] * D + d], v);
}

__global__ void self_rows(const float* __restrict__ selfsum, const float* __restrict__ cnt,
                          unsigned short* __restrict__ eattr_bf, int N, int D, int E) {
  long long i = (long long)blockIdx.x * blockDim.x + threadIdx.x;
  if (i >= (long long)N * D) return;
  int n = (int)(i / D);
  float c = fmaxf(cnt[n], 1.0f);
  eattr_bf[(size_t)E * D + i] = f2bf(selfsum[i] / c);
}

// per-(edge,head) attention logits + segment max (monotone float->uint key)
__global__ void att_logits(const float* __restrict__ xl, const float* __restrict__ xr,
                           const float* __restrict__ ebuf, const float* __restrict__ att,
                           const int* __restrict__ src, const int* __restrict__ dst,
                           float* __restrict__ logits, unsigned int* __restrict__ lmax,
                           int E, int Nn, int OC, int D) {
  int idx = blockIdx.x * blockDim.x + threadIdx.x;
  int total = (E + Nn) * 4;
  if (idx >= total) return;
  int h = idx & 3, e = idx >> 2;
  int sn, dn;
  if (e < E) { sn = src[e]; dn = dst[e]; } else { sn = dn = e - E; }
  const float* xlp = xl + (size_t)sn * D + h * OC;
  const float* xrp = xr + (size_t)dn * D + h * OC;
  const float* ep  = ebuf + (size_t)e * D + h * OC;
  const float* ah  = att + h * OC;
  float a = 0.0f;
  for (int c = 0; c < OC; ++c) {
    float m = xlp[c] + xrp[c] + ep[c];
    m = m > 0.0f ? m : 0.2f * m;          // leaky_relu(0.2)
    a += ah[c] * m;
  }
  logits[(size_t)e * 4 + h] = a;
  unsigned int k = __float_as_uint(a);
  k = (k & 0x80000000u) ? ~k : (k | 0x80000000u);
  atomicMax(&lmax[dn * 4 + h], k);
}

__global__ void att_exp(float* __restrict__ logits, const unsigned int* __restrict__ lmax,
                        const int* __restrict__ dst, float* __restrict__ denom,
                        int E, int Nn) {
  int idx = blockIdx.x * blockDim.x + threadIdx.x;
  int total = (E + Nn) * 4;
  if (idx >= total) return;
  int h = idx & 3, e = idx >> 2;
  int dn = (e < E) ? dst[e] : (e - E);
  unsigned int k = lmax[dn * 4 + h];
  unsigned int bits = (k & 0x80000000u) ? (k & 0x7FFFFFFFu) : ~k;
  float m = __uint_as_float(bits);
  float a = __expf(logits[(size_t)e * 4 + h] - m);
  logits[(size_t)e * 4 + h] = a;
  atomicAdd(&denom[dn * 4 + h], a);
}

__global__ void att_agg(const float* __restrict__ a, const float* __restrict__ denom,
                        const float* __restrict__ xl, const int* __restrict__ src,
                        const int* __restrict__ dst, float* __restrict__ out,
                        int E, int Nn, int OC, int D) {
  long long idx = (long long)blockIdx.x * blockDim.x + threadIdx.x;
  long long total = (long long)(E + Nn) * D;
  if (idx >= total) return;
  int e = (int)(idx / D), d = (int)(idx % D);
  int h = d / OC;
  int sn, dn;
  if (e < E) { sn = src[e]; dn = dst[e]; } else { sn = dn = e - E; }
  float alpha = a[(size_t)e * 4 + h] / fmaxf(denom[dn * 4 + h], 1e-16f);
  atomicAdd(&out[(size_t)dn * D + d], alpha * xl[(size_t)sn * D + d]);
}

__global__ void epilogue_concat(const float* __restrict__ acc, const float* __restrict__ bias,
                                float* __restrict__ xout, int N, int D) {
  long long i = (long long)blockIdx.x * blockDim.x + threadIdx.x;
  if (i >= (long long)N * D) return;
  int d = (int)(i % D);
  float v = acc[i] + bias[d];
  v = (v > 0.0f) ? v : (__expf(v) - 1.0f);   // ELU
  xout[i] = v;
}

__global__ void epilogue_final(const float* __restrict__ acc, const float* __restrict__ bias,
                               float* __restrict__ xout, int N) {
  long long i = (long long)blockIdx.x * blockDim.x + threadIdx.x;
  if (i >= (long long)N * 128) return;
  int n = (int)(i / 128), c = (int)(i % 128);
  const float* row = acc + (size_t)n * 512;
  float m = 0.25f * (row[c] + row[128 + c] + row[256 + c] + row[384 + c]);
  xout[i] = m + bias[c];
}

__global__ void pool_kernel(const float* __restrict__ x, const int* __restrict__ mask,
                            float* __restrict__ g, int N) {
  __shared__ float sp[256], sc[256], sm[256];
  int c = blockIdx.x;       // channel 0..127
  int t = threadIdx.x;
  float ap = 0.0f, ac_ = 0.0f, mp = 0.0f;
  for (int n = t; n < N; n += 256) {
    float v = x[(size_t)n * 128 + c];
    float mf = mask[n] ? 1.0f : 0.0f;
    ap += v * mf; ac_ += v * (1.0f - mf); mp += mf;
  }
  sp[t] = ap; sc[t] = ac_; sm[t] = mp;
  __syncthreads();
  for (int s = 128; s > 0; s >>= 1) {
    if (t < s) { sp[t] += sp[t + s]; sc[t] += sc[t + s]; sm[t] += sm[t + s]; }
    __syncthreads();
  }
  if (t == 0) {
    float np = fmaxf(sm[0], 1.0f);
    float nc = fmaxf((float)N - sm[0], 1.0f);
    g[c] = 0.8f * sp[0] / np + 0.2f * sc[0] / nc;
  }
}

__global__ void score_kernel(const float* __restrict__ g, const float* __restrict__ W1,
                             const float* __restrict__ b1, const float* __restrict__ W2,
                             const float* __restrict__ b2, float* __restrict__ out) {
  __shared__ float h[64];
  int t = threadIdx.x;      // 64 threads
  float a = b1[t];
  for (int i = 0; i < 128; ++i) a += g[i] * W1[i * 64 + t];
  h[t] = a > 0.0f ? a : 0.0f;
  __syncthreads();
  if (t == 0) {
    float s = b2[0];
    for (int i = 0; i < 64; ++i) s += h[i] * W2[i];
    out[0] = 1.0f / (1.0f + __expf(-s));
  }
}

// --------------------------------- driver -----------------------------------

extern "C" void kernel_launch(void* const* d_in, const int* in_sizes, int n_in,
                              void* d_out, int out_size, void* d_ws, size_t ws_size,
                              hipStream_t stream) {
  (void)in_sizes; (void)n_in; (void)out_size; (void)ws_size;
  const int N = 20000, E = 150000, ND = 384;
  const int EED = 388;       // logical edge-feature dim
  const int EAL = 392;       // padded stride (16B-aligned rows in bf16)
  const int E2 = E + N;

  const float* node_features = (const float*)d_in[0];
  const int*   edge_index    = (const int*)d_in[1];
  const float* edge_attr     = (const float*)d_in[2];
  const float* epr_scores    = (const float*)d_in[3];
  const int*   path_mask     = (const int*)d_in[5];
  const float* epr_W1 = (const float*)d_in[10];
  const float* epr_b1 = (const float*)d_in[11];
  const float* epr_W2 = (const float*)d_in[12];
  const float* epr_b2 = (const float*)d_in[13];
  const float* ps_W1  = (const float*)d_in[35];
  const float* ps_b1  = (const float*)d_in[36];
  const float* ps_W2  = (const float*)d_in[37];
  const float* ps_b2  = (const float*)d_in[38];

  struct L { const float *Wl, *bl, *Wr, *br, *We, *att, *bias; int INK, OUT, OC; };
  L layers[3] = {
    {(const float*)d_in[14], (const float*)d_in[15], (const float*)d_in[16],
     (const float*)d_in[17], (const float*)d_in[18], (const float*)d_in[19],
     (const float*)d_in[20], 384, 128, 32},
    {(const float*)d_in[21], (const float*)d_in[22], (const float*)d_in[23],
     (const float*)d_in[24], (const float*)d_in[25], (const float*)d_in[26],
     (const float*)d_in[27], 128, 128, 32},
    {(const float*)d_in[28], (const float*)d_in[29], (const float*)d_in[30],
     (const float*)d_in[31], (const float*)d_in[32], (const float*)d_in[33],
     (const float*)d_in[34], 128, 512, 128},
  };

  const int* src = edge_index;
  const int* dst = edge_index + E;

  // workspace bump allocator (all offsets 256B-aligned)
  char* ws = (char*)d_ws;
  size_t off = 0;
  auto alloc = [&](size_t bytes) -> char* {
    char* p = ws + off;
    off += (bytes + 255) & ~(size_t)255;
    return p;
  };
  float*          epr_feat = (float*)alloc((size_t)E * 4 * 4);
  unsigned short* eattr_bf = (unsigned short*)alloc((size_t)E2 * EAL * 2);
  float*          selfsum  = (float*)alloc((size_t)N * EAL * 4);
  float*          cnt      = (float*)alloc((size_t)N * 4);
  unsigned short* xb       = (unsigned short*)alloc((size_t)N * ND * 2);
  float*          xl       = (float*)alloc((size_t)N * 512 * 4);
  float*          xr       = (float*)alloc((size_t)N * 512 * 4);
  float*          ebuf     = (float*)alloc((size_t)E2 * 512 * 4);
  float*          aw       = (float*)alloc((size_t)E2 * 4 * 4);
  unsigned int*   lmax     = (unsigned int*)alloc((size_t)N * 4 * 4);
  float*          denom    = (float*)alloc((size_t)N * 4 * 4);
  float*          outacc   = (float*)alloc((size_t)N * 512 * 4);
  float*          xcur     = (float*)alloc((size_t)N * 128 * 4);

  float* out_score = (float*)d_out;
  float* out_x     = out_score + 1;
  float* out_g     = out_x + (size_t)N * 128;

  auto blocks = [](long long n) { return dim3((unsigned)((n + 255) / 256)); };

  // epr_feat MLP, edge feature assembly (bf16), self-loop mean rows
  epr_mlp<<<blocks(E), 256, 0, stream>>>(epr_scores, epr_W1, epr_b1, epr_W2, epr_b2,
                                         epr_feat, E);
  zero_f32<<<blocks((long long)N * EAL), 256, 0, stream>>>(selfsum, (long long)N * EAL);
  zero_f32<<<blocks(N), 256, 0, stream>>>(cnt, N);
  count_dst<<<blocks(E), 256, 0, stream>>>(dst, cnt, E);
  build_eattr<<<blocks((long long)E * EAL), 256, 0, stream>>>(
      edge_attr, epr_feat, dst, eattr_bf, selfsum, E, EAL);
  self_rows<<<blocks((long long)N * EAL), 256, 0, stream>>>(
      selfsum, cnt, eattr_bf, N, EAL, E);

  for (int l = 0; l < 3; ++l) {
    const L& P = layers[l];
    const float* xin = (l == 0) ? node_features : xcur;
    const int OUT = P.OUT, INK = P.INK, OC = P.OC;

    cvt_bf16<<<blocks((long long)N * INK), 256, 0, stream>>>(xin, xb, (long long)N * INK);

    dim3 gN(OUT / BN, (N + BM - 1) / BM);
    gemm_bf16_wmma<<<gN, 256, 0, stream>>>(xb, INK, P.Wl, OUT, P.bl, xl, OUT, N, OUT, INK);
    gemm_bf16_wmma<<<gN, 256, 0, stream>>>(xb, INK, P.Wr, OUT, P.br, xr, OUT, N, OUT, INK);
    dim3 gE(OUT / BN, (E2 + BM - 1) / BM);
    gemm_bf16_wmma<<<gE, 256, 0, stream>>>(eattr_bf, EAL, P.We, OUT, nullptr, ebuf, OUT,
                                           E2, OUT, EED);

    zero_f32<<<blocks((long long)N * 4), 256, 0, stream>>>((float*)lmax, (long long)N * 4);
    zero_f32<<<blocks((long long)N * 4), 256, 0, stream>>>(denom, (long long)N * 4);
    zero_f32<<<blocks((long long)N * OUT), 256, 0, stream>>>(outacc, (long long)N * OUT);

    att_logits<<<blocks((long long)E2 * 4), 256, 0, stream>>>(
        xl, xr, ebuf, P.att, src, dst, aw, lmax, E, N, OC, OUT);
    att_exp<<<blocks((long long)E2 * 4), 256, 0, stream>>>(aw, lmax, dst, denom, E, N);
    att_agg<<<blocks((long long)E2 * OUT), 256, 0, stream>>>(
        aw, denom, xl, src, dst, outacc, E, N, OC, OUT);

    if (l < 2) {
      epilogue_concat<<<blocks((long long)N * OUT), 256, 0, stream>>>(
          outacc, P.bias, xcur, N, OUT);
    } else {
      epilogue_final<<<blocks((long long)N * 128), 256, 0, stream>>>(
          outacc, P.bias, out_x, N);
    }
  }

  pool_kernel<<<128, 256, 0, stream>>>(out_x, path_mask, out_g, N);
  score_kernel<<<1, 64, 0, stream>>>(out_g, ps_W1, ps_b1, ps_W2, ps_b2, out_score);
}
